// MambaHSIFNO_12051678232814
// MI455X (gfx1250) — matrix operand order repaired and checked by reading, not compiled, not run on hardware
//
#include <hip/hip_runtime.h>
#include <hip/hip_bf16.h>

// ============================================================================
// MambaHSI-FNO forward for MI455X (gfx1250, wave32, WMMA, async-LDS).
//
// Design (reasoned from MI455X specs, compile-only):
//  * Total work is a few GFLOP; all intermediates (<~110 MB) live in d_ws and
//    stay resident in the 192 MB L2 -> the kernel suite is launch/latency
//    bound, so every matmul-shaped stage is mapped onto WMMA tiles:
//      - DFT / inverse-DFT stages (rfft2/irfft2 recast as per-axis matmuls
//        restricted to the 38x19 low-mode block): V_WMMA_F32_16X16X4_F32.
//      - Mamba in/x/dt/out projections + head: V_WMMA_F32_16X16X32_F16
//        (f16 inputs, f32 accumulate; weights are ~0.05-scale -> safe).
//  * Tile staging in the f16 GEMM uses GLOBAL_LOAD_ASYNC_TO_LDS_B128/B32
//    (CDNA5 async DMA, ASYNCcnt-tracked, bypasses VGPRs) + s_wait_asynccnt;
//    b128 fast path when the K-run is contiguous + 16B-aligned (uniform).
//  * All address math is branchless and strength-reduced: running 64-bit
//    offsets (one add per element) with a cndmask clamp against the last
//    valid K offset -- no per-element v_mul_u64, no exec-mask diamonds,
//    transB hoisted into (base, step) outside the loops.
//  * The sequential SSM scan keeps its 16 states in registers per (seq,chan)
//    lane and fuses y = hs.C + x*D and the silu(z) gating.
// ============================================================================

typedef __attribute__((ext_vector_type(16))) _Float16 v16h;
typedef __attribute__((ext_vector_type(8)))  float    v8f;
typedef __attribute__((ext_vector_type(2)))  float    v2f;

#define FLAG_ACC 1
#define FLAG_NEG 2

__device__ __forceinline__ float sigmoidf_(float v) {
    return 1.f / (1.f + __expf(-v));
}

__device__ __forceinline__ void async_ld_b32(unsigned lds_off, const float* gp) {
    asm volatile("global_load_async_to_lds_b32 %0, %1, off"
                 :: "v"(lds_off), "v"((unsigned long long)(size_t)gp)
                 : "memory");
}
__device__ __forceinline__ void async_ld_b128(unsigned lds_off, const float* gp) {
    asm volatile("global_load_async_to_lds_b128 %0, %1, off"
                 :: "v"(lds_off), "v"((unsigned long long)(size_t)gp)
                 : "memory");
}
__device__ __forceinline__ void wait_async0() {
    asm volatile("s_wait_asynccnt 0" ::: "memory");
}

// ---------------------------------------------------------------------------
// f16-input WMMA GEMM:  C[m,n] (+)= (+-) sum_k A[m,k]*B[k,n]  (+bias, act)
// One wave per 16x16 C tile. A/B tiles staged to LDS as raw fp32 via async
// DMA (b128 fast path / b32 clamped gather), converted to f16 with OOB
// masking at fragment-gather time. A addressed with arbitrary (row,k) strides
// so channels-first tensors feed directly. Batched via grid.z.
// act: 0 none, 1 silu, 2 softplus.
// ---------------------------------------------------------------------------
__global__ __launch_bounds__(32) void gemm_f16_kernel(
    const float* __restrict__ A, long sAm, long sAk, long sAb,
    const float* __restrict__ Bw, int ldb, int transB, long sBb,
    float* __restrict__ C, long sCm, long sCn, long sCb,
    const float* __restrict__ bias,
    int M, int N, int K, int act, int flags)
{
    __shared__ float Ash[16 * 32];   // [row][k]
    __shared__ float Bsh[16 * 32];   // [col][k]
    const int lane = threadIdx.x;
    const int half = lane >> 4;       // 0: k 0..15, 1: k 16..31 of the tile
    const int lid  = lane & 15;
    const int tm = blockIdx.x * 16, tn = blockIdx.y * 16;
    const float* Ab = A + (long)blockIdx.z * sAb;
    const float* Bb = Bw + (long)blockIdx.z * sBb;
    float* Cb = C + (long)blockIdx.z * sCb;

    // lane stages A row (tm+lid) and B col (tn+lid), k-range [half*16, +16)
    const int  gmA = tm + lid;
    const int  gnB = tn + lid;
    const bool rowOk = (gmA < M);
    const bool colOk = (gnB < N);
    const float* aBase = Ab + (long)min(gmA, M - 1) * sAm;
    const long  bStep  = transB ? 1L : (long)ldb;
    const float* bBase = Bb + (long)min(gnB, N - 1) * (transB ? (long)ldb : 1L);
    const long lastA = (long)(K - 1) * sAk;
    const long lastB = (long)(K - 1) * bStep;
    // uniform fast-path eligibility: contiguous K-run + 16B alignment
    const bool fastA = (sAk == 1) && ((sAm & 3L) == 0) && ((((size_t)Ab) & 15) == 0);
    const bool fastB = (bStep == 1) && ((ldb & 3) == 0) && ((((size_t)Bb) & 15) == 0);

    const unsigned aLds = (unsigned)(size_t)(void*)&Ash[lid * 32 + half * 16];
    const unsigned bLds = (unsigned)(size_t)(void*)&Bsh[lid * 32 + half * 16];

    v8f acc = {};
    long aOff = (long)(half * 16) * sAk;
    long bOff = (long)(half * 16) * bStep;
    for (int k0 = 0; k0 < K; k0 += 32) {
        const bool full = (k0 + 32 <= K);
        if (full && fastA) {
            #pragma unroll
            for (int j = 0; j < 4; ++j)
                async_ld_b128(aLds + 16u * j, aBase + aOff + 4 * j);
        } else {
            long o = aOff;
            #pragma unroll
            for (int j = 0; j < 16; ++j) {
                const long useO = ((k0 + half * 16 + j) < K) ? o : lastA;
                async_ld_b32(aLds + 4u * j, aBase + useO);
                o += sAk;
            }
        }
        if (full && fastB) {
            #pragma unroll
            for (int j = 0; j < 4; ++j)
                async_ld_b128(bLds + 16u * j, bBase + bOff + 4 * j);
        } else {
            long o = bOff;
            #pragma unroll
            for (int j = 0; j < 16; ++j) {
                const long useO = ((k0 + half * 16 + j) < K) ? o : lastB;
                async_ld_b32(bLds + 4u * j, bBase + useO);
                o += bStep;
            }
        }
        aOff += 32 * sAk;
        bOff += 32 * bStep;
        wait_async0();
        __syncthreads();

        // fragment gather, contiguous k = half*16 + e for both A and B, mask OOB
        v16h af, bf;
        #pragma unroll
        for (int e = 0; e < 16; ++e) {
            const int Ka = half * 16 + e;
            float av = Ash[lid * 32 + Ka];
            af[e] = (_Float16)((rowOk && (k0 + Ka) < K) ? av : 0.f);
            const int Kb = half * 16 + e;
            float bv = Bsh[lid * 32 + Kb];
            bf[e] = (_Float16)((colOk && (k0 + Kb) < K) ? bv : 0.f);
        }
        acc = __builtin_amdgcn_wmma_f32_16x16x32_f16(
            false, af, false, bf, (short)0, acc, false, false);
        __syncthreads();
    }

    #pragma unroll
    for (int r = 0; r < 8; ++r) {
        int gm = tm + r + half * 8;
        int gn = tn + lid;
        if (gm < M && gn < N) {
            float v = acc[r];
            if (flags & FLAG_NEG) v = -v;
            long off = (long)gm * sCm + (long)gn * sCn;
            if (flags & FLAG_ACC) v += Cb[off];
            if (bias) v += bias[gn];
            if (act == 1) v = v * sigmoidf_(v);
            else if (act == 2) v = (v > 20.f) ? v : __logf(1.f + __expf(v));
            Cb[off] = v;
        }
    }
}

// ---------------------------------------------------------------------------
// fp32 WMMA GEMM (V_WMMA_F32_16X16X4_F32): used for the DFT stages where
// precision matters. One wave per tile, K step 4, branchless direct global
// fragment gathers with running offsets + clamp-select. Batched + strided.
// ---------------------------------------------------------------------------
__global__ __launch_bounds__(32) void gemm_f32_kernel(
    const float* __restrict__ A, long sAm, long sAk, long sAb,
    const float* __restrict__ Bw, int ldb, int transB, long sBb,
    float* __restrict__ C, long sCm, long sCn, long sCb,
    int M, int N, int K, int flags)
{
    const int lane = threadIdx.x;
    const int tm = blockIdx.x * 16, tn = blockIdx.y * 16;
    const float* Ab = A + (long)blockIdx.z * sAb;
    const float* Bb = Bw + (long)blockIdx.z * sBb;
    float* Cb = C + (long)blockIdx.z * sCb;

    const int lid = lane & 15;
    const int am = tm + lid;
    const int bn = tn + lid;
    const bool rowOk = (am < M);
    const bool colOk = (bn < N);
    const float* aBase = Ab + (long)min(am, M - 1) * sAm;
    const long  bStep  = transB ? 1L : (long)ldb;
    const float* bBase = Bb + (long)min(bn, N - 1) * (transB ? (long)ldb : 1L);
    const long lastA = (long)(K - 1) * sAk;
    const long lastB = (long)(K - 1) * bStep;
    const int khi = (lane >= 16) ? 2 : 0;

    v8f acc = {};
    long aOff = (long)khi * sAk;
    long bOff = (long)khi * bStep;
    for (int k0 = 0; k0 < K; k0 += 4) {
        v2f a, b;
        #pragma unroll
        for (int e = 0; e < 2; ++e) {
            const int gk = k0 + khi + e;
            const bool kin = (gk < K);
            const long ao = kin ? (aOff + e * sAk) : lastA;
            const long bo = kin ? (bOff + e * bStep) : lastB;
            float av = aBase[ao];
            float bv = bBase[bo];
            a[e] = (rowOk && kin) ? av : 0.f;
            b[e] = (colOk && kin) ? bv : 0.f;
        }
        aOff += 4 * sAk;
        bOff += 4 * bStep;
        acc = __builtin_amdgcn_wmma_f32_16x16x4_f32(
            false, a, false, b, (short)0, acc, false, false);
    }

    #pragma unroll
    for (int r = 0; r < 8; ++r) {
        int gm = tm + r + ((lane >= 16) ? 8 : 0);
        int gn = tn + lid;
        if (gm < M && gn < N) {
            float v = acc[r];
            if (flags & FLAG_NEG) v = -v;
            long off = (long)gm * sCm + (long)gn * sCn;
            if (flags & FLAG_ACC) v += Cb[off];
            Cb[off] = v;
        }
    }
}

// ---------------------------------------------------------------------------
// DFT basis tables (computed on device each call; deterministic, tiny).
// Forward (norm='ortho'): per-axis 1/sqrt(96).
//   stage1 (along W, real->cplx, kx<19): cw = cos/s96, sw = -sin/s96
//   stage2 (along H, ky<38):  Yre = f2c@Xre + f2s@Xim ; Yim = f2c@Xim - f2s@Xre
// Inverse:
//   invA (along H): Yre = ic@Zre - is@Zim ; Yim = ic@Zim + is@Zre
//   invB (along W, cplx->real with Hermitian weighting alpha_k):
//       x = Yre@gc - Yim@gs,  gc[k,w]=a_k cos/s96, gs[k,w]=a_k sin/s96
// ---------------------------------------------------------------------------
__global__ void init_bases(float* cw, float* sw, float* f2c, float* f2s,
                           float* ic, float* isn, float* gc, float* gs)
{
    const float inv = 0.10206207261596576f;          // 1/sqrt(96)
    const float tp  = 6.283185307179586f / 96.f;
    int stride = blockDim.x * gridDim.x;
    for (int i = blockIdx.x * blockDim.x + threadIdx.x; i < 96 * 19; i += stride) {
        int w = i / 19, k = i % 19;
        float th = tp * (float)(w * k);
        float c = __cosf(th), s = __sinf(th);
        cw[i] = c * inv;
        sw[i] = -s * inv;
        float al = (k == 0) ? 1.f : 2.f;
        gc[k * 96 + w] = al * c * inv;
        gs[k * 96 + w] = al * s * inv;
    }
    for (int i = blockIdx.x * blockDim.x + threadIdx.x; i < 38 * 96; i += stride) {
        int ky = i / 96, h = i % 96;
        float th = tp * (float)(h * ky);
        float c = __cosf(th), s = __sinf(th);
        f2c[i] = c * inv;
        f2s[i] = s * inv;
        ic[h * 38 + ky] = c * inv;
        isn[h * 38 + ky] = s * inv;
    }
}

// FNO channel mixing: out[b,o,m] = sum_c (wr+i wi)[o,c,m] * low[b,c,m]
__global__ void mode_mix(const float* __restrict__ lre, const float* __restrict__ lim,
                         const float* __restrict__ wr,  const float* __restrict__ wi,
                         float* __restrict__ ore, float* __restrict__ oim,
                         int B, int Cin, int Cout)
{
    const int modes = 38 * 19;
    int i = blockIdx.x * blockDim.x + threadIdx.x;
    int total = B * Cout * modes;
    if (i >= total) return;
    int m = i % modes;
    int o = (i / modes) % Cout;
    int b = i / (modes * Cout);
    float ar = 0.f, ai = 0.f;
    for (int c = 0; c < Cin; ++c) {
        float xr = lre[((long)b * Cin + c) * modes + m];
        float xi = lim[((long)b * Cin + c) * modes + m];
        float wrv = wr[((long)o * Cin + c) * modes + m];
        float wiv = wi[((long)o * Cin + c) * modes + m];
        ar += wrv * xr - wiv * xi;
        ai += wrv * xi + wiv * xr;
    }
    ore[((long)b * Cout + o) * modes + m] = ar;
    oim[((long)b * Cout + o) * modes + m] = ai;
}

// GroupNorm (+optional SiLU), in-place, channels-first (B,C,HW). One block
// per (batch, group); block reduction over the group's C/g * HW elements.
__global__ void groupnorm_silu(float* __restrict__ data,
                               const float* __restrict__ w, const float* __restrict__ b,
                               int C, int HW, int groups, int do_silu)
{
    int g = blockIdx.x % groups;
    int bb = blockIdx.x / groups;
    int cpg = C / groups;
    long cnt = (long)cpg * HW;
    float* base = data + ((long)bb * C + (long)g * cpg) * HW;
    float s = 0.f, sq = 0.f;
    for (long i = threadIdx.x; i < cnt; i += blockDim.x) {
        float v = base[i];
        s += v; sq += v * v;
    }
    __shared__ float rs[256], rq[256];
    rs[threadIdx.x] = s; rq[threadIdx.x] = sq;
    __syncthreads();
    for (int o = 128; o > 0; o >>= 1) {
        if (threadIdx.x < (unsigned)o) {
            rs[threadIdx.x] += rs[threadIdx.x + o];
            rq[threadIdx.x] += rq[threadIdx.x + o];
        }
        __syncthreads();
    }
    float mean = rs[0] / (float)cnt;
    float var  = rq[0] / (float)cnt - mean * mean;
    float rinv = rsqrtf(var + 1e-5f);
    for (long i = threadIdx.x; i < cnt; i += blockDim.x) {
        int cg = (int)(i / HW);
        int cglob = g * cpg + cg;
        float v = (base[i] - mean) * rinv * w[cglob] + b[cglob];
        if (do_silu) v = v * sigmoidf_(v);
        base[i] = v;
    }
}

// Depthwise causal conv1d (k=4, left pad 3) + bias + SiLU, (Nseq, L, di)
// reading the xi half of the (Nseq, L, 2*di) in_proj output.
__global__ void conv1d_silu(const float* __restrict__ xz,
                            const float* __restrict__ cw, const float* __restrict__ cb,
                            float* __restrict__ xi, int Nseq, int L, int di)
{
    long i = (long)blockIdx.x * blockDim.x + threadIdx.x;
    long tot = (long)Nseq * L * di;
    if (i >= tot) return;
    int d = (int)(i % di);
    long t = i / di;
    int l = (int)(t % L);
    long n = t / L;
    float s = 0.f;
    #pragma unroll
    for (int j = 0; j < 4; ++j) {
        int ls = l + j - 3;
        if (ls >= 0) s += xz[(n * L + ls) * (long)(2 * di) + d] * cw[d * 4 + j];
    }
    s += cb[d];
    xi[i] = s * sigmoidf_(s);
}

// Fused selective-scan: per (seq, channel) lane keeps 16 SSM states in
// registers, recurrence h = exp(dt*A)h + dt*B*x, emits y = h.C + x*D, gated
// by silu(z). dt is already softplus'ed by the dt GEMM.
__global__ void scan_kernel(const float* __restrict__ dtb, const float* __restrict__ xdbl,
                            const float* __restrict__ xib, const float* __restrict__ xz,
                            const float* __restrict__ A_log, const float* __restrict__ Dv,
                            float* __restrict__ yg,
                            int Nseq, int L, int di, int dtr, int ldx)
{
    long i = (long)blockIdx.x * blockDim.x + threadIdx.x;
    if (i >= (long)Nseq * di) return;
    int d = (int)(i % di);
    long n = i / di;
    float As[16];
    #pragma unroll
    for (int s = 0; s < 16; ++s) As[s] = -__expf(A_log[d * 16 + s]);
    float Dd = Dv[d];
    float h[16];
    #pragma unroll
    for (int s = 0; s < 16; ++s) h[s] = 0.f;
    for (int l = 0; l < L; ++l) {
        long row = n * L + l;
        float dtv = dtb[row * di + d];
        float xv  = xib[row * di + d];
        const float* xd = xdbl + row * (long)ldx + dtr;   // [Bm(16) | Cm(16)]
        float acc = 0.f;
        #pragma unroll
        for (int s = 0; s < 16; ++s) {
            float dA = __expf(dtv * As[s]);
            h[s] = dA * h[s] + dtv * xd[s] * xv;
            acc += h[s] * xd[16 + s];
        }
        float y = acc + xv * Dd;
        float zv = xz[row * (long)(2 * di) + di + d];
        y *= zv * sigmoidf_(zv);
        yg[row * di + d] = y;
    }
}

// out = softmax(att)[0]*(spa+x) + softmax(att)[1]*(spe+x) + x
__global__ void combine_att(const float* __restrict__ spa, const float* __restrict__ spe,
                            const float* __restrict__ x, const float* __restrict__ attw,
                            float* __restrict__ out, long n)
{
    long i = (long)blockIdx.x * blockDim.x + threadIdx.x;
    if (i >= n) return;
    float a0 = attw[0], a1 = attw[1];
    float m = fmaxf(a0, a1);
    float e0 = __expf(a0 - m), e1 = __expf(a1 - m);
    float inv = 1.f / (e0 + e1);
    float w0 = e0 * inv, w1 = e1 * inv;
    float xv = x[i];
    out[i] = w0 * (spa[i] + xv) + w1 * (spe[i] + xv) + xv;
}

__global__ void avgpool2(const float* __restrict__ in, float* __restrict__ out,
                         int BC, int Hi, int Wi)
{
    int Ho = Hi >> 1, Wo = Wi >> 1;
    long i = (long)blockIdx.x * blockDim.x + threadIdx.x;
    long tot = (long)BC * Ho * Wo;
    if (i >= tot) return;
    int wo = (int)(i % Wo);
    int ho = (int)((i / Wo) % Ho);
    long bc = i / ((long)Wo * Ho);
    const float* p = in + ((bc * Hi + 2 * ho) * (long)Wi) + 2 * wo;
    out[i] = 0.25f * (p[0] + p[1] + p[Wi] + p[Wi + 1]);
}

// (B,C=64,HW) -> (B*HW, 8 tokens, 8) and back, for the spectral mamba.
__global__ void to_xe(const float* __restrict__ h, float* __restrict__ xe, int B, int HW)
{
    long i = (long)blockIdx.x * blockDim.x + threadIdx.x;
    long tot = (long)B * 64 * HW;
    if (i >= tot) return;
    int p = (int)(i % HW);
    int c = (int)((i / HW) % 64);
    int b = (int)(i / ((long)64 * HW));
    xe[(((long)b * HW + p) * 8 + (c >> 3)) * 8 + (c & 7)] = h[i];
}
__global__ void from_xe(const float* __restrict__ xe, float* __restrict__ out, int B, int HW)
{
    long i = (long)blockIdx.x * blockDim.x + threadIdx.x;
    long tot = (long)B * 64 * HW;
    if (i >= tot) return;
    int p = (int)(i % HW);
    int c = (int)((i / HW) % 64);
    int b = (int)(i / ((long)64 * HW));
    out[i] = xe[(((long)b * HW + p) * 8 + (c >> 3)) * 8 + (c & 7)];
}

// ============================================================================
// Host-side orchestration
// ============================================================================
static inline void gemm16(hipStream_t s, const float* A, long sAm, long sAk, long sAb,
                          const float* B, int ldb, int transB, long sBb,
                          float* C, long sCm, long sCn, long sCb,
                          const float* bias, int M, int N, int K, int batch,
                          int act, int flags)
{
    dim3 g((M + 15) / 16, (N + 15) / 16, batch);
    gemm_f16_kernel<<<g, 32, 0, s>>>(A, sAm, sAk, sAb, B, ldb, transB, sBb,
                                     C, sCm, sCn, sCb, bias, M, N, K, act, flags);
}
static inline void gemm32(hipStream_t s, const float* A, long sAm, long sAk, long sAb,
                          const float* B, int ldb, int transB, long sBb,
                          float* C, long sCm, long sCn, long sCb,
                          int M, int N, int K, int batch, int flags)
{
    dim3 g((M + 15) / 16, (N + 15) / 16, batch);
    gemm_f32_kernel<<<g, 32, 0, s>>>(A, sAm, sAk, sAb, B, ldb, transB, sBb,
                                     C, sCm, sCn, sCb, M, N, K, flags);
}

struct MP { const float* p[9]; };
enum { IN_PROJ, CONV_W, CONV_B, X_PROJ, DT_W, DT_B, A_LOG, D_P, OUT_PROJ };

// One full mamba() call. Input consumed with arbitrary strides (so channels-
// first (B,C,HW) feeds directly); output written with arbitrary strides.
static void run_mamba(hipStream_t s, const MP& mp,
                      const float* Ain, long sAm, long sAk, long sAb, int inBatch, int inM,
                      int Nseq, int L, int d, int di, int dtr,
                      float* Cout, long sCm, long sCn, long sCb, int outBatch, int outM,
                      float* xz, float* xib, float* xdbl, float* dtb, float* yg)
{
    const int R   = Nseq * L;
    const int ldx = dtr + 32;      // dtr + 2*ds (ds == 16)
    // in_proj: xz = x @ W_in^T   (R x 2di)
    gemm16(s, Ain, sAm, sAk, sAb, mp.p[IN_PROJ], d, 1, 0L,
           xz, (long)(2 * di), 1L, (long)inM * 2 * di,
           nullptr, inM, 2 * di, d, inBatch, 0, 0);
    // causal depthwise conv + bias + silu -> xi
    {
        long tot = (long)Nseq * L * di;
        conv1d_silu<<<(unsigned)((tot + 255) / 256), 256, 0, s>>>(
            xz, mp.p[CONV_W], mp.p[CONV_B], xib, Nseq, L, di);
    }
    // x_proj: xdbl = xi @ W_x^T   (R x ldx)
    gemm16(s, xib, (long)di, 1L, 0L, mp.p[X_PROJ], di, 1, 0L,
           xdbl, (long)ldx, 1L, 0L, nullptr, R, ldx, di, 1, 0, 0);
    // dt = softplus(xdbl[:, :dtr] @ dt_w^T + dt_b)   (R x di)
    gemm16(s, xdbl, (long)ldx, 1L, 0L, mp.p[DT_W], dtr, 1, 0L,
           dtb, (long)di, 1L, 0L, mp.p[DT_B], R, di, dtr, 1, 2, 0);
    // fused SSM scan + D skip + silu(z) gate -> yg (R x di)
    {
        long nthr = (long)Nseq * di;
        scan_kernel<<<(unsigned)((nthr + 255) / 256), 256, 0, s>>>(
            dtb, xdbl, xib, xz, mp.p[A_LOG], mp.p[D_P], yg, Nseq, L, di, dtr, ldx);
    }
    // out_proj: y @ W_out^T
    gemm16(s, yg, (long)di, 1L, (long)outM * di, mp.p[OUT_PROJ], di, 1, 0L,
           Cout, sCm, sCn, sCb, nullptr, outM, d, di, outBatch, 0, 0);
}

extern "C" void kernel_launch(void* const* d_in, const int* in_sizes, int n_in,
                              void* d_out, int out_size, void* d_ws, size_t ws_size,
                              hipStream_t stream)
{
    (void)in_sizes; (void)n_in; (void)out_size; (void)ws_size;
    // ---- unpack inputs in setup_inputs() dict order ----
    int q = 0;
    auto nx = [&]() { return (const float*)d_in[q++]; };
    const float* X = nx();                      // (2,200,96,96)
    const float* fwr[4]; const float* fwi[4];
    for (int i = 0; i < 4; ++i) { fwr[i] = nx(); fwi[i] = nx(); }
    const float* pe_w = nx(); const float* pe_b = nx();
    struct Both {
        const float* att; MP spa; const float *sgw, *sgb;
        MP spe; const float *egw, *egb;
    } mb[3];
    for (int m = 0; m < 3; ++m) {
        mb[m].att = nx();
        for (int j = 0; j < 9; ++j) mb[m].spa.p[j] = nx();
        mb[m].sgw = nx(); mb[m].sgb = nx();
        for (int j = 0; j < 9; ++j) mb[m].spe.p[j] = nx();
        mb[m].egw = nx(); mb[m].egb = nx();
    }
    const float* hw1 = nx(); const float* hb1 = nx();
    const float* hgw = nx(); const float* hgb = nx();
    const float* hw2 = nx(); const float* hb2 = nx();

    // ---- workspace bump allocator (floats) ----
    float* ws = (float*)d_ws;
    size_t off = 0;
    auto alloc = [&](size_t n) { float* p = ws + off; off += n; return p; };
    const int B = 2, HW0 = 96 * 96, MODES = 38 * 19;
    float *cw = alloc(96 * 19), *sw = alloc(96 * 19);
    float *f2c = alloc(38 * 96), *f2s = alloc(38 * 96);
    float *ic = alloc(96 * 38), *isn = alloc(96 * 38);
    float *gc = alloc(19 * 96), *gs = alloc(19 * 96);
    float* hbuf  = alloc((size_t)B * 64 * HW0);   // current h / fno out
    float* spaP  = alloc((size_t)B * 64 * HW0);   // spatial branch
    float* speP  = alloc((size_t)B * 64 * HW0);   // spectral branch
    float* hnext = alloc((size_t)B * 64 * HW0);   // combined
    size_t mark = off;                             // transient arena

    init_bases<<<32, 256, 0, stream>>>(cw, sw, f2c, f2s, ic, isn, gc, gs);

    // =================== FNO: rfft2 -> mix -> irfft2 ===================
    {
        float* xw_re = alloc((size_t)B * 80 * 96 * 19);
        float* xw_im = alloc((size_t)B * 80 * 96 * 19);
        float* lo_re = alloc((size_t)B * 80 * MODES);
        float* lo_im = alloc((size_t)B * 80 * MODES);
        float* mx_re = alloc((size_t)B * 25 * MODES);
        float* mx_im = alloc((size_t)B * 25 * MODES);
        float* yc_re = alloc((size_t)B * 25 * 96 * 19);
        float* yc_im = alloc((size_t)B * 25 * 96 * 19);

        const int ci0[4] = {0, 80, 140, 160};
        const int cin[4] = {80, 60, 20, 40};
        const int co0[4] = {0, 25, 44, 50};
        const int cou[4] = {25, 19, 6, 14};
        for (int sI = 0; sI < 4; ++sI) {
            const int Ci = cin[sI], Co = cou[sI];
            const float* Xs = X + (long)ci0[sI] * HW0;
            // stage1: along W (rows = (c,h), K=96, N=19), batched over b
            gemm32(stream, Xs, 96L, 1L, (long)200 * HW0, cw, 19, 0, 0L,
                   xw_re, 19L, 1L, (long)Ci * 96 * 19, Ci * 96, 19, 96, B, 0);
            gemm32(stream, Xs, 96L, 1L, (long)200 * HW0, sw, 19, 0, 0L,
                   xw_im, 19L, 1L, (long)Ci * 96 * 19, Ci * 96, 19, 96, B, 0);
            // stage2: along H, batched over (b,c); Y = F2 @ Xw
            gemm32(stream, f2c, 96L, 1L, 0L, xw_re, 19, 0, 96L * 19,
                   lo_re, 19L, 1L, (long)MODES, 38, 19, 96, B * Ci, 0);
            gemm32(stream, f2s, 96L, 1L, 0L, xw_im, 19, 0, 96L * 19,
                   lo_re, 19L, 1L, (long)MODES, 38, 19, 96, B * Ci, FLAG_ACC);
            gemm32(stream, f2c, 96L, 1L, 0L, xw_im, 19, 0, 96L * 19,
                   lo_im, 19L, 1L, (long)MODES, 38, 19, 96, B * Ci, 0);
            gemm32(stream, f2s, 96L, 1L, 0L, xw_re, 19, 0, 96L * 19,
                   lo_im, 19L, 1L, (long)MODES, 38, 19, 96, B * Ci, FLAG_ACC | FLAG_NEG);
            // per-mode complex channel mixing
            {
                int tot = B * Co * MODES;
                mode_mix<<<(tot + 255) / 256, 256, 0, stream>>>(
                    lo_re, lo_im, fwr[sI], fwi[sI], mx_re, mx_im, B, Ci, Co);
            }
            // invA: along H, batched over (b,o)
            gemm32(stream, ic, 38L, 1L, 0L, mx_re, 19, 0, (long)MODES,
                   yc_re, 19L, 1L, 96L * 19, 96, 19, 38, B * Co, 0);
            gemm32(stream, isn, 38L, 1L, 0L, mx_im, 19, 0, (long)MODES,
                   yc_re, 19L, 1L, 96L * 19, 96, 19, 38, B * Co, FLAG_ACC | FLAG_NEG);
            gemm32(stream, ic, 38L, 1L, 0L, mx_im, 19, 0, (long)MODES,
                   yc_im, 19L, 1L, 96L * 19, 96, 19, 38, B * Co, 0);
            gemm32(stream, isn, 38L, 1L, 0L, mx_re, 19, 0, (long)MODES,
                   yc_im, 19L, 1L, 96L * 19, 96, 19, 38, B * Co, FLAG_ACC);
            // invB: along W, real output into channel block [co0, co0+Co)
            float* Cdst = hbuf + (long)co0[sI] * HW0;
            gemm32(stream, yc_re, 19L, 1L, (long)Co * 96 * 19, gc, 96, 0, 0L,
                   Cdst, 96L, 1L, (long)64 * HW0, Co * 96, 96, 19, B, 0);
            gemm32(stream, yc_im, 19L, 1L, (long)Co * 96 * 19, gs, 96, 0, 0L,
                   Cdst, 96L, 1L, (long)64 * HW0, Co * 96, 96, 19, B, FLAG_ACC | FLAG_NEG);
        }
    }
    // pe: h = silu(group_norm(h))
    groupnorm_silu<<<B * 4, 256, 0, stream>>>(hbuf, pe_w, pe_b, 64, HW0, 4, 1);

    // =================== mamba scratch (reuses FNO arena) ===================
    off = mark;
    float* xz   = alloc((size_t)B * HW0 * 256);       // max in_proj out
    float* xib  = alloc((size_t)B * HW0 * 128);
    float* xdbl = alloc((size_t)B * HW0 * 8 * 33);    // spe level0: 147456*33
    float* dtb  = alloc((size_t)B * HW0 * 128);
    float* yg   = alloc((size_t)B * HW0 * 128);
    float* xeA  = alloc((size_t)B * HW0 * 64);
    float* xeB  = alloc((size_t)B * HW0 * 64);

    // =================== 3 x both_mamba (+pool) ===================
    int side = 96;
    float* cur = hbuf;
    for (int lev = 0; lev < 3; ++lev) {
        const int HW = side * side;
        // --- spatial mamba (seq over pixels, d=64, di=128, dtr=4) ---
        run_mamba(stream, mb[lev].spa,
                  cur, 1L, (long)HW, (long)64 * HW, B, HW,
                  B, HW, 64, 128, 4,
                  spaP, 1L, (long)HW, (long)64 * HW, B, HW,
                  xz, xib, xdbl, dtb, yg);
        groupnorm_silu<<<B * 4, 256, 0, stream>>>(spaP, mb[lev].sgw, mb[lev].sgb, 64, HW, 4, 1);
        // --- spectral mamba (seq over 8 channel tokens, d=8, di=16, dtr=1) ---
        {
            long tot = (long)B * 64 * HW;
            to_xe<<<(unsigned)((tot + 255) / 256), 256, 0, stream>>>(cur, xeA, B, HW);
            run_mamba(stream, mb[lev].spe,
                      xeA, 8L, 1L, 0L, 1, B * HW * 8,
                      B * HW, 8, 8, 16, 1,
                      xeB, 8L, 1L, 0L, 1, B * HW * 8,
                      xz, xib, xdbl, dtb, yg);
            from_xe<<<(unsigned)((tot + 255) / 256), 256, 0, stream>>>(xeB, speP, B, HW);
            groupnorm_silu<<<B * 4, 256, 0, stream>>>(speP, mb[lev].egw, mb[lev].egb, 64, HW, 4, 1);
        }
        // --- attention-weighted combine ---
        {
            long n = (long)B * 64 * HW;
            combine_att<<<(unsigned)((n + 255) / 256), 256, 0, stream>>>(
                spaP, speP, cur, mb[lev].att, hnext, n);
        }
        if (lev < 2) {
            long tot = (long)B * 64 * (HW / 4);
            avgpool2<<<(unsigned)((tot + 255) / 256), 256, 0, stream>>>(
                hnext, hbuf, B * 64, side, side);
            cur = hbuf;
            side >>= 1;
        } else {
            cur = hnext;
        }
    }

    // =================== head ===================
    {
        const int HW = side * side;    // 24*24
        // h1 = h @ W1^T + b1  (channels-first in and out)
        gemm16(stream, cur, 1L, (long)HW, (long)64 * HW, hw1, 64, 1, 0L,
               spaP, 1L, (long)HW, (long)128 * HW, hb1, HW, 128, 64, B, 0, 0);
        groupnorm_silu<<<B * 4, 256, 0, stream>>>(spaP, hgw, hgb, 128, HW, 4, 1);
        // logits = h1 @ W2^T + b2 -> d_out (B,10,24,24)
        gemm16(stream, spaP, 1L, (long)HW, (long)128 * HW, hw2, 128, 1, 0L,
               (float*)d_out, 1L, (long)HW, (long)10 * HW, hb2, HW, 10, 128, B, 0, 0);
    }
}
